// SpecificProfile_18270790877667
// MI455X (gfx1250) — matrix operand order repaired and checked by reading, not compiled, not run on hardware
//
#include <hip/hip_runtime.h>
#include <hip/hip_bf16.h>
#include <math.h>

typedef __attribute__((ext_vector_type(16))) _Float16 v16h;
typedef __attribute__((ext_vector_type(8)))  _Float16 v8h;
typedef __attribute__((ext_vector_type(2)))  _Float16 v2h;
typedef __attribute__((ext_vector_type(8)))  float    v8f;

#define KLEN   20
#define ALPHA  21
#define APAD   24           // alphabet padded so 8-half groups are 16B aligned
#define UPROF  600
#define NFRM   6
#define TILE   334          // positions per tile
#define LOUT   315          // TILE - KLEN + 1
#define NBATCH 768          // 16*8*6
#define KPAD   480          // KLEN*APAD -> 15 chunks of 32
#define NKCH   15
#define XPAD   8160         // 340 rows * 24 (covers m+k up to 338, zero-padded)
#define UG     32           // profiles per block (2 WMMA n-tiles)
#define NUGRP  19           // ceil(600/32)

#define S_ELEMS (16*8*UPROF)          // 76800
#define R_ELEMS (KLEN*ALPHA*UPROF)    // 252000

// ---------------- S init: fill with -inf ----------------
__global__ void sp_init_s(float* S) {
    int i = blockIdx.x * 256 + threadIdx.x;
    if (i < S_ELEMS) S[i] = -INFINITY;
}

// ---------------- R = log(max(P/Q, eps)), fp32, layout [k*21+a][u] ----------------
__global__ void sp_rprep(const float* __restrict__ P, const float* __restrict__ Q,
                         float* __restrict__ R) {
    int i = blockIdx.x * 256 + threadIdx.x;
    if (i < R_ELEMS) {
        int a = (i / UPROF) % ALPHA;
        float v = P[i] / Q[a];
        R[i] = logf(fmaxf(v, 1e-6f));
    }
}

// ---------------- main: batched GEMM via v_wmma_f32_16x16x32_f16 ----------------
__global__ __launch_bounds__(128)
void sp_conv_wmma(const float* __restrict__ X,   // (768, 334, 21) fp32 one-hot
                  const float* __restrict__ Rf,  // (420, 600) fp32 log-odds
                  float* __restrict__ Z,         // (768, 315, 600)
                  float* __restrict__ S) {       // (128, 600), pre-set to -inf
    // X row, f16, position stride APAD=24, cols 21-23 and rows >=334 are zero
    __shared__ __align__(16) _Float16 lds_x[XPAD];
    // R tile pre-swizzled into B-fragment order:
    // [c*2+ut][lane][16 halves = K kb..kb+7, 16+kb..16+kb+7], 32B per lane
    __shared__ __align__(16) _Float16 lds_rs[NKCH * 2 * 32 * 16];

    const int b   = blockIdx.x;     // batch (t*8+n)*6+f
    const int ug  = blockIdx.y;     // u-group
    const int tid = threadIdx.x;
    const int u0  = ug * UG;

    // --- stage X row into LDS (f16, stride 24), branch-free row copy ---
    const float* xb = X + (size_t)b * (TILE * ALPHA);
    for (int m = tid; m < TILE; m += 128) {
        const float* xr = xb + m * ALPHA;
        _Float16*    xd = &lds_x[m * APAD];
        #pragma unroll
        for (int j = 0; j < ALPHA; ++j) xd[j] = (_Float16)xr[j];
        xd[21] = (_Float16)0.0f; xd[22] = (_Float16)0.0f; xd[23] = (_Float16)0.0f;
    }
    for (int i = TILE * APAD + tid; i < XPAD; i += 128)
        lds_x[i] = (_Float16)0.0f;

    // --- stage R tile into LDS in swizzled fragment order (paired-half writes) ---
    for (int w = tid; w < NKCH * 2 * 32 * 8; w += 128) {
        int s2   = w & 7;            // which half-pair within the lane's 16
        int lane = (w >> 3) & 31;
        int cu   = w >> 8;           // c*2 + ut
        int c    = cu >> 1;
        int ut   = cu & 1;
        int n    = lane & 15;
        int kb   = (lane >> 4) << 3; // 0 or 8
        int u    = u0 + ut * 16 + n;
        int kf0  = c * 32 + (s2 >> 2) * 16 + kb + 2 * (s2 & 3);   // flattened k*24+a index
        float f0 = 0.0f, f1 = 0.0f;
        if (u < UPROF) {
            int k0 = kf0 / APAD, a0 = kf0 - k0 * APAD;
            if (a0 < ALPHA) f0 = Rf[(k0 * ALPHA + a0) * UPROF + u];
            int kf1 = kf0 + 1;
            int k1 = kf1 / APAD, a1 = kf1 - k1 * APAD;
            if (a1 < ALPHA) f1 = Rf[(k1 * ALPHA + a1) * UPROF + u];
        }
        v2h pr; pr[0] = (_Float16)f0; pr[1] = (_Float16)f1;
        *(v2h*)&lds_rs[(cu * 32 + lane) * 16 + 2 * s2] = pr;
    }
    __syncthreads();

    const int wave   = tid >> 5;
    const int lane   = tid & 31;
    const int lane16 = lane & 15;
    const int mhalf  = lane >> 4;   // A: K-offset half; C/D: M-offset half

    float smax[2] = {-INFINITY, -INFINITY};

    // each wave owns m-tiles {wave, wave+4, ...} -> 5 tiles
    for (int mt = wave; mt < 20; mt += 4) {
        // hoist all 15 A fragments for this m-tile: 2x ds_load_b128 each
        v16h afrag[NKCH];
        const int abase = (mt * 16 + lane16) * APAD + mhalf * 8;
        #pragma unroll
        for (int c = 0; c < NKCH; ++c) {
            v8h lo = *(const v8h*)&lds_x[abase + c * 32];        // K = kb+0..7
            v8h hi = *(const v8h*)&lds_x[abase + c * 32 + 16];   // K = 16+kb+0..7
            afrag[c] = __builtin_shufflevector(lo, hi,
                0, 1, 2, 3, 4, 5, 6, 7, 8, 9, 10, 11, 12, 13, 14, 15);
        }

        #pragma unroll
        for (int ut = 0; ut < 2; ++ut) {
            v8f acc = {0.f, 0.f, 0.f, 0.f, 0.f, 0.f, 0.f, 0.f};
            #pragma unroll
            for (int c = 0; c < NKCH; ++c) {
                // whole B fragment: one 32B aligned LDS read (2x ds_load_b128)
                v16h bfrag = *(const v16h*)&lds_rs[((c * 2 + ut) * 32 + lane) * 16];
                acc = __builtin_amdgcn_wmma_f32_16x16x32_f16(
                    false, afrag[c], false, bfrag, (short)0, acc, false, false);
            }

            // C/D layout: row m = mt*16 + mhalf*8 + r, col u = u0 + ut*16 + lane16
            const int u = u0 + ut * 16 + lane16;
            const bool uval = (u < UPROF);
            float lmax = -INFINITY;
            #pragma unroll
            for (int r = 0; r < 8; ++r) {
                int m = mt * 16 + mhalf * 8 + r;
                if (m < LOUT) {
                    float zv = acc[r];
                    lmax = fmaxf(lmax, zv);
                    if (uval) Z[((size_t)b * LOUT + m) * UPROF + u] = zv;
                }
            }
            // fold the two M-halves (lane L <-> L+16 share the same u)
            lmax = fmaxf(lmax, __shfl_xor(lmax, 16, 32));
            smax[ut] = fmaxf(smax[ut], lmax);
        }
    }

    // S[t*8+n][u] = max over f and l -> atomic max across 6 frames & 4 waves
    const int tn = b / NFRM;
    if (lane < 16) {
        #pragma unroll
        for (int ut = 0; ut < 2; ++ut) {
            int u = u0 + ut * 16 + lane;
            if (u < UPROF) atomicMax(&S[tn * UPROF + u], smax[ut]);
        }
    }
}

extern "C" void kernel_launch(void* const* d_in, const int* in_sizes, int n_in,
                              void* d_out, int out_size, void* d_ws, size_t ws_size,
                              hipStream_t stream) {
    const float* X = (const float*)d_in[0];   // (16,8,6,334,21) fp32
    const float* P = (const float*)d_in[1];   // (20,21,600) fp32
    const float* Q = (const float*)d_in[2];   // (21,) fp32

    float* out = (float*)d_out;
    float* S = out;                        // 76800
    float* R = out + S_ELEMS;              // 252000
    float* Z = out + S_ELEMS + R_ELEMS;    // 768*315*600

    sp_init_s<<<(S_ELEMS + 255) / 256, 256, 0, stream>>>(S);
    sp_rprep<<<(R_ELEMS + 255) / 256, 256, 0, stream>>>(P, Q, R);
    dim3 grid(NBATCH, NUGRP);
    sp_conv_wmma<<<grid, 128, 0, stream>>>(X, R, Z, S);
}
